// HinTop_35493609734227
// MI455X (gfx1250) — matrix operand order repaired and checked by reading, not compiled, not run on hardware
//
#include <hip/hip_runtime.h>

// Problem constants (from reference): B=64, H=12, P=197, D=64, top-k=10, T=1.0
#define BH    768                 // B*H
#define PP    197
#define DD    64
#define MT    13                  // ceil(197/16) M-tiles
#define KPAD  224                 // 197 padded to 7*32 for K
#define NBLK  (BH * MT)           // 9984 blocks
#define NUMK  10
#define INV_N (1.0f / 9682944.0f) // 1 / (64*12*197*64)

typedef _Float16 half8 __attribute__((ext_vector_type(8)));
typedef _Float16 v16h  __attribute__((ext_vector_type(16)));
typedef float    v8f   __attribute__((ext_vector_type(8)));

union FragU { v16h v; half8 h8[2]; };

__device__ __forceinline__ float rmax8(float v) {
    for (int off = 4; off; off >>= 1) v = fmaxf(v, __shfl_xor(v, off, 8));
    return v;
}
__device__ __forceinline__ float rsum8(float v) {
    for (int off = 4; off; off >>= 1) v += __shfl_xor(v, off, 8);
    return v;
}

__global__ __launch_bounds__(128)
void HinTop_topk_attend_mse_kernel(const float* __restrict__ att_s,
                                   const float* __restrict__ att_t,
                                   const float* __restrict__ v_s,
                                   const float* __restrict__ v_t,
                                   float* __restrict__ partials)
{
    __shared__ __align__(16) _Float16 Als[16][KPAD];   // masked/normalized softmax tile
    __shared__ __align__(16) _Float16 Vls[DD][KPAD];   // V transposed: [n][k]
    __shared__ float red[128];

    const int tid  = threadIdx.x;
    const int lane = tid & 31;
    const int wave = tid >> 5;            // 0..3 -> N-tile
    const bool hi  = (lane >= 16);
    const int bh   = blockIdx.x / MT;
    const int mt   = blockIdx.x % MT;

    const float* atts[2] = { att_s + (size_t)bh * PP * PP,
                             att_t + (size_t)bh * PP * PP };
    const float* vsrc[2] = { v_s + (size_t)bh * PP * DD,
                             v_t + (size_t)bh * PP * DD };

    // Zero LDS once: valid region is fully overwritten each stream; padding stays 0.
    {
        uint32_t* az = (uint32_t*)&Als[0][0];
        uint32_t* vz = (uint32_t*)&Vls[0][0];
        for (int i = tid; i < 16 * KPAD / 2; i += 128) az[i] = 0u;
        for (int i = tid; i < DD * KPAD / 2; i += 128) vz[i] = 0u;
    }
    __syncthreads();

    v8f cres[2];

    #pragma unroll 2            // s becomes compile-time: no dyn-indexed cres (no v_movreld)
    for (int s = 0; s < 2; ++s) {
        if (s) __syncthreads();           // previous stream's WMMA reads done

        // ---- Phase 1: per-row softmax + top-10 mask + renorm -> Als (f16) ----
        {
            const int row = tid >> 3;     // 16 rows, 8 threads each
            const int rl  = tid & 7;
            const int mg  = mt * 16 + row;
            if (mg < PP) {
                const float* arow = atts[s] + (size_t)mg * PP;
                if (s == 0) {
                    // warm L2/GL1 for stream 1's logit row (global_prefetch_b8)
                    __builtin_prefetch(atts[1] + (size_t)mg * PP + rl * 25, 0, 1);
                }
                float r[25];
                float mx = -3.0e38f;
                #pragma unroll
                for (int j = 0; j < 25; ++j) {
                    int idx = rl + (j << 3);
                    r[j] = (idx < PP) ? arow[idx] : -3.0e38f;
                    mx = fmaxf(mx, r[j]);
                }
                mx = rmax8(mx);
                // 10 select-max passes: bound ends as the 10th-largest logit
                float bound = 3.0e38f;
                for (int it = 0; it < NUMK; ++it) {
                    float cur = -3.0e38f;
                    #pragma unroll
                    for (int j = 0; j < 25; ++j)
                        if (r[j] < bound) cur = fmaxf(cur, r[j]);
                    bound = rmax8(cur);
                }
                // masked exp + normalization folded into the A tile
                float ssum = 0.f;
                #pragma unroll
                for (int j = 0; j < 25; ++j) {
                    float e = (r[j] >= bound) ? __expf(r[j] - mx) : 0.f;
                    r[j] = e;
                    ssum += e;
                }
                ssum = rsum8(ssum);
                const float inv = 1.f / ssum;
                #pragma unroll
                for (int j = 0; j < 25; ++j) {
                    int idx = rl + (j << 3);
                    if (idx < PP) Als[row][idx] = (_Float16)(r[j] * inv);
                }
            }
        }

        // ---- Phase 2: stage V transposed in LDS (f16), coalesced global reads ----
        {
            const float* vp = vsrc[s];
            for (int t = tid; t < PP * DD; t += 128) {
                int q = t >> 6, d = t & 63;
                Vls[d][q] = (_Float16)vp[t];
            }
        }
        __syncthreads();

        // ---- Phase 3: WMMA, 1 M-tile x 4 N-tiles (one per wave), K = 224 ----
        {
            const int m = lane & 15;
            const int n = (wave << 4) + (lane & 15);
            v8f c = {0.f, 0.f, 0.f, 0.f, 0.f, 0.f, 0.f, 0.f};
            #pragma unroll
            for (int ks = 0; ks < KPAD / 32; ++ks) {
                const int kb = ks * 32;
                FragU a, b;
                // A 16x32 f16 layout: halves 0-7 = K [hi*8, +8); 8-15 = K [16+hi*8, +8)
                a.h8[0] = *(const half8*)&Als[m][kb + (hi ? 8 : 0)];
                a.h8[1] = *(const half8*)&Als[m][kb + 16 + (hi ? 8 : 0)];
                // B 32x16 f16 layout: halves 0-7 = K [hi*8, +8); 8-15 = K [16+hi*8, +8), lane = N
                b.h8[0] = *(const half8*)&Vls[n][kb + (hi ? 8 : 0)];
                b.h8[1] = *(const half8*)&Vls[n][kb + 16 + (hi ? 8 : 0)];
                c = __builtin_amdgcn_wmma_f32_16x16x32_f16(
                        false, a.v, false, b.v, (short)0, c, false, false);
            }
            cres[s] = c;
        }
    }

    // ---- squared diff of the two streams' C tiles, block reduction ----
    float part = 0.f;
    #pragma unroll
    for (int rr = 0; rr < 8; ++rr) {
        int mg = mt * 16 + rr + (hi ? 8 : 0);   // C/D layout: M = vgpr + hi*8
        if (mg < PP) {
            float d = cres[0][rr] - cres[1][rr];
            part += d * d;
        }
    }
    red[tid] = part;
    __syncthreads();
    for (int st = 64; st > 0; st >>= 1) {
        if (tid < st) red[tid] += red[tid + st];
        __syncthreads();
    }
    if (tid == 0) partials[blockIdx.x] = red[0];
}

__global__ __launch_bounds__(256)
void HinTop_mse_finalize_kernel(const float* __restrict__ partials,
                                float* __restrict__ out)
{
    __shared__ float red[256];
    float s = 0.f;
    for (int i = threadIdx.x; i < NBLK; i += 256) s += partials[i];
    red[threadIdx.x] = s;
    __syncthreads();
    for (int st = 128; st > 0; st >>= 1) {
        if (threadIdx.x < st) red[threadIdx.x] += red[threadIdx.x + st];
        __syncthreads();
    }
    if (threadIdx.x == 0) out[0] = red[0] * INV_N;
}

extern "C" void kernel_launch(void* const* d_in, const int* in_sizes, int n_in,
                              void* d_out, int out_size, void* d_ws, size_t ws_size,
                              hipStream_t stream) {
    (void)in_sizes; (void)n_in; (void)out_size; (void)ws_size;
    const float* att_s = (const float*)d_in[0];
    const float* att_t = (const float*)d_in[1];
    const float* v_s   = (const float*)d_in[2];
    const float* v_t   = (const float*)d_in[3];
    float* partials = (float*)d_ws;           // NBLK floats of scratch
    float* out      = (float*)d_out;          // single f32 scalar

    HinTop_topk_attend_mse_kernel<<<NBLK, 128, 0, stream>>>(
        att_s, att_t, v_s, v_t, partials);
    HinTop_mse_finalize_kernel<<<1, 256, 0, stream>>>(partials, out);
}
